// SE3Transformer_25778393710946
// MI455X (gfx1250) — matrix operand order repaired and checked by reading, not compiled, not run on hardware
//
#include <hip/hip_runtime.h>
#include <hip/hip_bf16.h>

typedef __attribute__((ext_vector_type(16))) _Float16 v16h;
typedef __attribute__((ext_vector_type(8)))  _Float16 v8h;
typedef __attribute__((ext_vector_type(8)))  float    v8f;

#define P_    128
#define B_    16
#define N_    2048          // B_*P_
#define EPG_  16256         // 127*128
#define E_    260096        // B_*EPG_

__device__ __host__ constexpr int moff_(int l) { return l == 0 ? 0 : (l == 1 ? 1 : (l == 2 ? 4 : 9)); }

// ---------------- WMMA helpers (CDNA5 16x16x32 f16 -> f32) ----------------

__device__ inline v8f wmma32f16(v16h a, v16h b, v8f c) {
  return __builtin_amdgcn_wmma_f32_16x16x32_f16(false, a, false, b, (short)0, c, false, false);
}

// A tile: 16 rows x 32 K, f16 row-major (ld in halfs).
// lane: row = lane&15, half = lane>>4; elem j: K = j + 8*half + (j>=8 ? 8 : 0)
__device__ inline v16h ldA16(const _Float16* base, int ld) {
  const int lane = threadIdx.x & 31;
  const _Float16* p = base + (size_t)(lane & 15) * ld + (lane >> 4) * 8;
  v8h lo = *(const v8h*)p;
  v8h hi = *(const v8h*)(p + 16);
  v16h a;
#pragma unroll
  for (int j = 0; j < 8; ++j) { a[j] = lo[j]; a[j + 8] = hi[j]; }
  return a;
}

// A tile from f32 source (converted to f16 on load)
__device__ inline v16h ldA32(const float* base, int ld) {
  const int lane = threadIdx.x & 31;
  const float* p = base + (size_t)(lane & 15) * ld + (lane >> 4) * 8;
  v16h a;
#pragma unroll
  for (int j = 0; j < 8; ++j) { a[j] = (_Float16)p[j]; a[j + 8] = (_Float16)p[j + 16]; }
  return a;
}

// B fragment pre-shuffled into register layout: frag[lane][j], one 32B read per lane
__device__ inline v16h ldBfrag(const _Float16* frag) {
  return *(const v16h*)(frag + (size_t)(threadIdx.x & 31) * 16);
}

// D store: row = r + 8*half, col = lane&15
__device__ inline void stD(float* base, int ld, v8f d) {
  const int lane = threadIdx.x & 31;
  float* p = base + (size_t)((lane >> 4) * 8) * ld + (lane & 15);
#pragma unroll
  for (int r = 0; r < 8; ++r) p[(size_t)r * ld] = d[r];
}

// ---------------- weight shuffling into B-frag order ----------------
// dst per matrix: [kb][nb][lane(32)][j(16)]; element = src[kb*32 + (j&7) + 8*(lane>>4) + 16*(j>>3)][(lane&15)+16*nb]

__global__ void k_mk_bfrag(_Float16* dst, const float* src, int Kt, int ncols, int nmat) {
  int i = blockIdx.x * blockDim.x + threadIdx.x;
  int per = Kt * ncols;
  if (i >= nmat * per) return;
  int m = i / per, rem = i % per;
  int j = rem & 15;
  int lane = (rem >> 4) & 31;
  int blk = rem >> 9;
  int nbc = ncols >> 4;
  int kb = blk / nbc, nb = blk % nbc;
  int K = kb * 32 + (j & 7) + 8 * (lane >> 4) + 16 * (j >> 3);
  int c = (lane & 15) + nb * 16;
  dst[i] = (_Float16)src[(size_t)m * per + (size_t)K * ncols + c];
}

// src [16x32] matrices -> frag of zero-padded [32x32]
__global__ void k_mk_bfrag_pad16(_Float16* dst, const float* src, int nmat) {
  int i = blockIdx.x * blockDim.x + threadIdx.x;
  if (i >= nmat * 1024) return;
  int m = i >> 10, rem = i & 1023;
  int j = rem & 15, lane = (rem >> 4) & 31, nb = rem >> 9;
  int K = (j & 7) + 8 * (lane >> 4) + 16 * (j >> 3);
  int c = (lane & 15) + nb * 16;
  dst[i] = (K < 16) ? (_Float16)src[m * 512 + K * 32 + c] : (_Float16)0.f;
}

// src [64x15] -> frag of zero-padded [64x16]
__global__ void k_mk_bfrag_wout(_Float16* dst, const float* src) {
  int i = blockIdx.x * blockDim.x + threadIdx.x;
  if (i >= 1024) return;
  int j = i & 15, lane = (i >> 4) & 31, kb = i >> 9;
  int K = kb * 32 + (j & 7) + 8 * (lane >> 4) + 16 * (j >> 3);
  int c = lane & 15;
  dst[i] = (c < 15) ? (_Float16)src[K * 15 + c] : (_Float16)0.f;
}

__global__ void k_zero16(_Float16* d, int n) {
  int i = blockIdx.x * blockDim.x + threadIdx.x;
  if (i < n) d[i] = (_Float16)0.f;
}

// ---------------- geometry: spherical harmonics + RBF ----------------

__global__ void k_geom(const float* x, const int* src, const int* dst,
                       float* ysph, _Float16* rbf16) {
  int e = blockIdx.x * blockDim.x + threadIdx.x;
  if (e >= E_) return;
  int s = src[e], t = dst[e];
  float rx = x[t * 3 + 0] - x[s * 3 + 0];
  float ry = x[t * 3 + 1] - x[s * 3 + 1];
  float rz = x[t * 3 + 2] - x[s * 3 + 2];
  float d = sqrtf(rx * rx + ry * ry + rz * rz + 1e-12f);
  float inv = 1.f / d;
  float X = rx * inv, Y = ry * inv, Z = rz * inv;
  float* yo = ysph + (size_t)e * 16;
  yo[0]  = 0.28209479177f;
  yo[1]  = 0.4886025119f * Y;
  yo[2]  = 0.4886025119f * Z;
  yo[3]  = 0.4886025119f * X;
  yo[4]  = 1.09254843059f * X * Y;
  yo[5]  = 1.09254843059f * Y * Z;
  yo[6]  = 0.31539156525f * (3.f * Z * Z - 1.f);
  yo[7]  = 1.09254843059f * X * Z;
  yo[8]  = 0.54627421529f * (X * X - Y * Y);
  yo[9]  = 0.59004358992f * Y * (3.f * X * X - Y * Y);
  yo[10] = 2.89061144264f * X * Y * Z;
  yo[11] = 0.45704579946f * Y * (5.f * Z * Z - 1.f);
  yo[12] = 0.37317633259f * Z * (5.f * Z * Z - 3.f);
  yo[13] = 0.45704579946f * X * (5.f * Z * Z - 1.f);
  yo[14] = 1.44530572132f * Z * (X * X - Y * Y);
  yo[15] = 0.59004358992f * X * (X * X - 3.f * Y * Y);
  _Float16* ro = rbf16 + (size_t)e * 16;
#pragma unroll
  for (int j = 0; j < 16; ++j) {
    float c = (float)j * (4.f / 15.f);
    float u = d - c;
    ro[j] = (_Float16)__expf(-2.f * u * u);
  }
}

// ---------------- q = h0 @ W   ([Nx32]@[32xoutc]) ----------------

__global__ void k_node_gemm_h0(const _Float16* h16, const _Float16* Wfrag, int outc, float* q) {
  int n0 = blockIdx.x * 16;
  v16h A = ldA16(h16 + (size_t)n0 * 512, 512);   // m=0 row of each node
  for (int nb = 0; nb < (outc >> 4); ++nb) {
    v8f c = {};
    c = wmma32f16(A, ldBfrag(Wfrag + nb * 512), c);
    stD(q + (size_t)n0 * outc + nb * 16, outc, c);
  }
}

// ---------------- fused per-dst layer kernel ----------------
// one 32-thread wave per dst node; 8 tiles of 16 candidate src points (a=0..127)

__global__ void k_layer_dst(const _Float16* h16, const float* q,
                            const float* ysph, const _Float16* rbf16,
                            const _Float16* w1p, const float* b1,
                            const _Float16* Wk, const _Float16* rW,
                            const _Float16* sW, const _Float16* Wv,
                            float* agg) {
  __shared__ _Float16 rh_s[128 * 32];
  __shared__ float logit_s[128];
  __shared__ float alpha_s[128];
  __shared__ float ys_s[128 * 16];
  __shared__ float h0p_s[128 * 32];
  __shared__ float agg_s[16 * 32];

  const int lane = threadIdx.x & 31;
  const int col = lane & 15, half = lane >> 4;
  const int nd = blockIdx.x;              // dst node
  const int g = nd >> 7, b = nd & 127;
  const long ebase = (long)g * EPG_;

  for (int i = lane; i < 16 * 32; i += 32) agg_s[i] = 0.f;
  for (int i = lane; i < 128 * 16; i += 32) {
    int a = i >> 4, j = i & 15;
    long e = (a == b) ? ebase : (ebase + a * 127 + (b - (int)(b > a)));
    ys_s[i] = ysph[e * 16 + j];
  }
  for (int i = lane; i < 128 * 32; i += 32) {
    int a = i >> 5, d = i & 31;
    h0p_s[i] = (float)h16[((size_t)(g * 128 + a) * 16) * 32 + d] + 1.0f;
  }
  __syncthreads();

  // ---- pass 1: rh, k, logits ----
  v16h Bw1[2], Bwk[2];
#pragma unroll
  for (int nb = 0; nb < 2; ++nb) {
    Bw1[nb] = ldBfrag(w1p + nb * 512);
    Bwk[nb] = ldBfrag(Wk + nb * 512);
  }
  float qv0 = q[(size_t)nd * 32 + col];
  float qv1 = q[(size_t)nd * 32 + col + 16];
  for (int t = 0; t < 8; ++t) {
    int aA = t * 16 + col;                // A-layout row index
    long e = (aA == b) ? ebase : (ebase + aA * 127 + (b - (int)(b > aA)));
    const _Float16* rp = rbf16 + e * 16 + half * 8;
    v8h lo = *(const v8h*)rp;
    v16h A;
#pragma unroll
    for (int j = 0; j < 8; ++j) { A[j] = lo[j]; A[j + 8] = (_Float16)0.f; }
#pragma unroll
    for (int nb = 0; nb < 2; ++nb) {
      v8f z = {};
      v8f cc = wmma32f16(A, Bw1[nb], z);
      int dcol = col + nb * 16;
      float bias = b1[dcol];
#pragma unroll
      for (int r = 0; r < 8; ++r)
        rh_s[(t * 16 + r + 8 * half) * 32 + dcol] = (_Float16)fmaxf(cc[r] + bias, 0.f);
    }
    __syncthreads();
    v16h rA = ldA16(rh_s + t * 16 * 32, 32);
    float part[8];
#pragma unroll
    for (int r = 0; r < 8; ++r) part[r] = 0.f;
#pragma unroll
    for (int nb = 0; nb < 2; ++nb) {
      v8f z = {};
      v8f kk = wmma32f16(rA, Bwk[nb], z);
      float qv = nb ? qv1 : qv0;
#pragma unroll
      for (int r = 0; r < 8; ++r) part[r] += kk[r] * qv;
    }
#pragma unroll
    for (int r = 0; r < 8; ++r) {
      float s = part[r];
      s += __shfl_xor(s, 8, 32);
      s += __shfl_xor(s, 4, 32);
      s += __shfl_xor(s, 2, 32);
      s += __shfl_xor(s, 1, 32);
      int row = t * 16 + r + 8 * half;
      if (col == 0) logit_s[row] = (row == b) ? -1e30f : s * 0.17677669529663687f;
    }
  }
  __syncthreads();

  // ---- softmax over the 127 valid edges ----
  float l0 = logit_s[lane], l1 = logit_s[lane + 32], l2 = logit_s[lane + 64], l3 = logit_s[lane + 96];
  float mx = fmaxf(fmaxf(l0, l1), fmaxf(l2, l3));
  for (int m = 16; m >= 1; m >>= 1) mx = fmaxf(mx, __shfl_xor(mx, m, 32));
  float e0 = __expf(l0 - mx), e1 = __expf(l1 - mx), e2 = __expf(l2 - mx), e3 = __expf(l3 - mx);
  float sm = e0 + e1 + e2 + e3;
  for (int m = 16; m >= 1; m >>= 1) sm += __shfl_xor(sm, m, 32);
  float isn = 1.f / sm;
  alpha_s[lane] = e0 * isn; alpha_s[lane + 32] = e1 * isn;
  alpha_s[lane + 64] = e2 * isn; alpha_s[lane + 96] = e3 * isn;
  __syncthreads();

  // ---- pass 2: R,S,v and alpha-weighted aggregation ----
  for (int l = 0; l < 4; ++l) {
    const int mo = moff_(l), ms = 2 * l + 1;
    v16h BrW[2], BsW[2], BWv[2];
#pragma unroll
    for (int nb = 0; nb < 2; ++nb) {
      BrW[nb] = ldBfrag(rW + l * 1024 + nb * 512);
      BsW[nb] = ldBfrag(sW + l * 1024 + nb * 512);
      BWv[nb] = ldBfrag(Wv + l * 1024 + nb * 512);
    }
    for (int t = 0; t < 8; ++t) {
      v16h rA = ldA16(rh_s + t * 16 * 32, 32);
      v8f R[2], S[2];
#pragma unroll
      for (int nb = 0; nb < 2; ++nb) {
        v8f z = {};
        R[nb] = wmma32f16(rA, BrW[nb], z);
        v8f z2 = {};
        S[nb] = wmma32f16(rA, BsW[nb], z2);
      }
      int nodeA = g * 128 + t * 16 + col;   // A-layout row -> src node
      if (t < 7)  // prefetch next tile's h rows (gfx1250 global_prefetch)
        __builtin_prefetch(h16 + ((size_t)(nodeA + 16) * 16 + mo) * 32, 0, 1);
      for (int m = 0; m < ms; ++m) {
        const _Float16* hp = h16 + ((size_t)nodeA * 16 + mo + m) * 32 + half * 8;
        v8h hlo = *(const v8h*)hp;
        v8h hhi = *(const v8h*)(hp + 16);
        v16h hA;
#pragma unroll
        for (int j = 0; j < 8; ++j) { hA[j] = hlo[j]; hA[j + 8] = hhi[j]; }
#pragma unroll
        for (int nb = 0; nb < 2; ++nb) {
          v8f z = {};
          v8f vD = wmma32f16(hA, BWv[nb], z);
          int d = col + nb * 16;
          float csum = 0.f;
#pragma unroll
          for (int r = 0; r < 8; ++r) {
            int row = t * 16 + r + 8 * half;
            float val = vD[r] * R[nb][r] + S[nb][r] * h0p_s[row * 32 + d] * ys_s[row * 16 + mo + m];
            csum += val * alpha_s[row];
          }
          csum += __shfl_xor(csum, 16, 32);
          if (half == 0) agg_s[(mo + m) * 32 + d] += csum;
        }
      }
    }
  }
  __syncthreads();
  for (int i = lane; i < 512; i += 32) agg[(size_t)nd * 512 + i] = agg_s[i];
}

// ---------------- node update: hn = h@Wskip + agg@Wo, norm-gate ----------------

__global__ void k_node_update(const _Float16* h16in, const float* agg,
                              const _Float16* Wskip, const _Float16* Wo,
                              const float* gw, const float* gb,
                              _Float16* h16out) {
  __shared__ float hn_s[16 * 16 * 32];   // [node][m][d]
  const int n0 = blockIdx.x * 16;
  const int lane = threadIdx.x & 31;
  const int col = lane & 15, half = lane >> 4;
  for (int l = 0; l < 4; ++l) {
    const int mo = moff_(l), ms = 2 * l + 1;
    v16h Bs[2], Bo[2];
#pragma unroll
    for (int nb = 0; nb < 2; ++nb) {
      Bs[nb] = ldBfrag(Wskip + l * 1024 + nb * 512);
      Bo[nb] = ldBfrag(Wo + l * 1024 + nb * 512);
    }
    for (int m = 0; m < ms; ++m) {
      v16h hA = ldA16(h16in + ((size_t)n0 * 16 + mo + m) * 32, 16 * 32);
      v16h aA = ldA32(agg + ((size_t)n0 * 16 + mo + m) * 32, 16 * 32);
#pragma unroll
      for (int nb = 0; nb < 2; ++nb) {
        v8f c = {};
        c = wmma32f16(hA, Bs[nb], c);
        c = wmma32f16(aA, Bo[nb], c);
#pragma unroll
        for (int r = 0; r < 8; ++r)
          hn_s[((r + 8 * half) * 16 + mo + m) * 32 + col + nb * 16] = c[r];
      }
    }
  }
  __syncthreads();
  for (int idx = lane; idx < 16 * 32; idx += 32) {
    int nl = idx >> 5, d = idx & 31;
    for (int l = 0; l < 4; ++l) {
      int mo = moff_(l), ms = 2 * l + 1;
      float s2 = 0.f;
      for (int m = 0; m < ms; ++m) {
        float v = hn_s[(nl * 16 + mo + m) * 32 + d];
        s2 += v * v;
      }
      float nrm = sqrtf(s2 + 1e-12f);
      float phi = fmaxf(nrm * gw[l * 32 + d] + gb[l * 32 + d], 0.f);
      float sc = phi / (nrm + 1e-6f);
      for (int m = 0; m < ms; ++m)
        h16out[((size_t)(n0 + nl) * 16 + mo + m) * 32 + d] =
            (_Float16)(hn_s[(nl * 16 + mo + m) * 32 + d] * sc);
    }
  }
}

// ---------------- fused per-dst output head ----------------

__global__ void k_out_dst(const _Float16* h16, const float* q64,
                          const float* ysph, const _Float16* rbf16,
                          const _Float16* w1op, const float* b1o,
                          const _Float16* WkO, const _Float16* roW,
                          const _Float16* WvO, float* aggO) {
  __shared__ _Float16 rh_s[128 * 32];
  __shared__ float logit_s[128];
  __shared__ float alpha_s[128];
  __shared__ float ys_s[128 * 16];
  __shared__ float agg_s[64];

  const int lane = threadIdx.x & 31;
  const int col = lane & 15, half = lane >> 4;
  const int nd = blockIdx.x;
  const int g = nd >> 7, b = nd & 127;
  const long ebase = (long)g * EPG_;

  for (int i = lane; i < 64; i += 32) agg_s[i] = 0.f;
  for (int i = lane; i < 128 * 16; i += 32) {
    int a = i >> 4, j = i & 15;
    long e = (a == b) ? ebase : (ebase + a * 127 + (b - (int)(b > a)));
    ys_s[i] = ysph[e * 16 + j];
  }
  __syncthreads();

  // ---- pass 1 ----
  v16h Bw1[2];
#pragma unroll
  for (int nb = 0; nb < 2; ++nb) Bw1[nb] = ldBfrag(w1op + nb * 512);
  float qv[4];
#pragma unroll
  for (int nb = 0; nb < 4; ++nb) qv[nb] = q64[(size_t)nd * 64 + col + nb * 16];
  for (int t = 0; t < 8; ++t) {
    int aA = t * 16 + col;
    long e = (aA == b) ? ebase : (ebase + aA * 127 + (b - (int)(b > aA)));
    const _Float16* rp = rbf16 + e * 16 + half * 8;
    v8h lo = *(const v8h*)rp;
    v16h A;
#pragma unroll
    for (int j = 0; j < 8; ++j) { A[j] = lo[j]; A[j + 8] = (_Float16)0.f; }
#pragma unroll
    for (int nb = 0; nb < 2; ++nb) {
      v8f z = {};
      v8f cc = wmma32f16(A, Bw1[nb], z);
      int dcol = col + nb * 16;
      float bias = b1o[dcol];
#pragma unroll
      for (int r = 0; r < 8; ++r)
        rh_s[(t * 16 + r + 8 * half) * 32 + dcol] = (_Float16)fmaxf(cc[r] + bias, 0.f);
    }
    __syncthreads();
    v16h rA = ldA16(rh_s + t * 16 * 32, 32);
    float part[8];
#pragma unroll
    for (int r = 0; r < 8; ++r) part[r] = 0.f;
    for (int nb = 0; nb < 4; ++nb) {
      v8f z = {};
      v8f kk = wmma32f16(rA, ldBfrag(WkO + nb * 512), z);
#pragma unroll
      for (int r = 0; r < 8; ++r) part[r] += kk[r] * qv[nb];
    }
#pragma unroll
    for (int r = 0; r < 8; ++r) {
      float s = part[r];
      s += __shfl_xor(s, 8, 32);
      s += __shfl_xor(s, 4, 32);
      s += __shfl_xor(s, 2, 32);
      s += __shfl_xor(s, 1, 32);
      int row = t * 16 + r + 8 * half;
      if (col == 0) logit_s[row] = (row == b) ? -1e30f : s * 0.125f;
    }
  }
  __syncthreads();

  float l0 = logit_s[lane], l1 = logit_s[lane + 32], l2 = logit_s[lane + 64], l3 = logit_s[lane + 96];
  float mx = fmaxf(fmaxf(l0, l1), fmaxf(l2, l3));
  for (int m = 16; m >= 1; m >>= 1) mx = fmaxf(mx, __shfl_xor(mx, m, 32));
  float e0 = __expf(l0 - mx), e1 = __expf(l1 - mx), e2 = __expf(l2 - mx), e3 = __expf(l3 - mx);
  float sm = e0 + e1 + e2 + e3;
  for (int m = 16; m >= 1; m >>= 1) sm += __shfl_xor(sm, m, 32);
  float isn = 1.f / sm;
  alpha_s[lane] = e0 * isn; alpha_s[lane + 32] = e1 * isn;
  alpha_s[lane + 64] = e2 * isn; alpha_s[lane + 96] = e3 * isn;
  __syncthreads();

  // ---- pass 2: m0 = sum_l (cc_l @ WvO_l) * (rh @ roW_l) ----
  for (int t = 0; t < 8; ++t) {
    v16h rA = ldA16(rh_s + t * 16 * 32, 32);
    v8f m0[4];
#pragma unroll
    for (int nb = 0; nb < 4; ++nb) { v8f z = {}; m0[nb] = z; }
    int nodeA = g * 128 + t * 16 + col;
    if (t < 7)
      __builtin_prefetch(h16 + ((size_t)(nodeA + 16) * 16) * 32, 0, 1);
    for (int l = 0; l < 4; ++l) {
      const int mo = moff_(l), ms = 2 * l + 1;
      float accLo[8], accHi[8];
#pragma unroll
      for (int j = 0; j < 8; ++j) { accLo[j] = 0.f; accHi[j] = 0.f; }
      for (int m = 0; m < ms; ++m) {
        const _Float16* hp = h16 + ((size_t)nodeA * 16 + mo + m) * 32 + half * 8;
        v8h hlo = *(const v8h*)hp;
        v8h hhi = *(const v8h*)(hp + 16);
        float yv = ys_s[(t * 16 + col) * 16 + mo + m];
#pragma unroll
        for (int j = 0; j < 8; ++j) {
          accLo[j] += (float)hlo[j] * yv;
          accHi[j] += (float)hhi[j] * yv;
        }
      }
      v16h cA;
#pragma unroll
      for (int j = 0; j < 8; ++j) { cA[j] = (_Float16)accLo[j]; cA[j + 8] = (_Float16)accHi[j]; }
      for (int nb = 0; nb < 4; ++nb) {
        v8f z = {};
        v8f rho = wmma32f16(rA, ldBfrag(roW + l * 2048 + nb * 512), z);
        v8f z2 = {};
        v8f cd = wmma32f16(cA, ldBfrag(WvO + l * 2048 + nb * 512), z2);
#pragma unroll
        for (int r = 0; r < 8; ++r) m0[nb][r] += cd[r] * rho[r];
      }
    }
#pragma unroll
    for (int nb = 0; nb < 4; ++nb) {
      float csum = 0.f;
#pragma unroll
      for (int r = 0; r < 8; ++r) {
        int row = t * 16 + r + 8 * half;
        csum += m0[nb][r] * alpha_s[row];
      }
      csum += __shfl_xor(csum, 16, 32);
      if (half == 0) agg_s[col + nb * 16] += csum;
    }
  }
  __syncthreads();
  for (int i = lane; i < 64; i += 32) aggO[(size_t)nd * 64 + i] = agg_s[i];
}

// ---------------- hout = aggO @ WoO + h0 @ WskipO ----------------

__global__ void k_hout(const float* aggO, const _Float16* h16,
                       const _Float16* WoO, const _Float16* WskipO,
                       float* hout) {
  int n0 = blockIdx.x * 16;
  v16h aLo = ldA32(aggO + (size_t)n0 * 64, 64);
  v16h aHi = ldA32(aggO + (size_t)n0 * 64 + 32, 64);
  v16h h0A = ldA16(h16 + (size_t)n0 * 512, 512);
  for (int nb = 0; nb < 4; ++nb) {
    v8f c = {};
    c = wmma32f16(aLo, ldBfrag(WoO + nb * 512), c);
    c = wmma32f16(aHi, ldBfrag(WoO + (4 + nb) * 512), c);
    c = wmma32f16(h0A, ldBfrag(WskipO + nb * 512), c);
    stD(hout + (size_t)n0 * 64 + nb * 16, 64, c);
  }
}

// ---------------- mean-pool + MLP head ----------------

__global__ void k_pool_mlp(const float* hout, const _Float16* Whid16, const float* bhid,
                           const _Float16* WoutP, const float* bout, float* out) {
  __shared__ float pooled_s[16 * 64];
  __shared__ float hid_s[16 * 64];
  const int lane = threadIdx.x & 31;
  const int col = lane & 15, half = lane >> 4;
  for (int g = 0; g < 16; ++g)
    for (int cc = lane; cc < 64; cc += 32) {
      float s = 0.f;
      for (int n = 0; n < 128; ++n) s += hout[(size_t)(g * 128 + n) * 64 + cc];
      pooled_s[g * 64 + cc] = s * (1.f / 128.f);
    }
  __syncthreads();
  v16h A0 = ldA32(pooled_s, 64);
  v16h A1 = ldA32(pooled_s + 32, 64);
  for (int nb = 0; nb < 4; ++nb) {
    v8f c = {};
    c = wmma32f16(A0, ldBfrag(Whid16 + nb * 512), c);
    c = wmma32f16(A1, ldBfrag(Whid16 + (4 + nb) * 512), c);
    float bb = bhid[col + nb * 16];
#pragma unroll
    for (int r = 0; r < 8; ++r)
      hid_s[(r + 8 * half) * 64 + col + nb * 16] = fmaxf(c[r] + bb, 0.f);
  }
  __syncthreads();
  v16h H0 = ldA32(hid_s, 64);
  v16h H1 = ldA32(hid_s + 32, 64);
  v8f c = {};
  c = wmma32f16(H0, ldBfrag(WoutP), c);
  c = wmma32f16(H1, ldBfrag(WoutP + 512), c);
  if (col < 15) {
    float bb = bout[col];
#pragma unroll
    for (int r = 0; r < 8; ++r) out[(r + 8 * half) * 15 + col] = c[r] + bb;
  }
}

// ---------------- host launcher ----------------

extern "C" void kernel_launch(void* const* d_in, const int* in_sizes, int n_in,
                              void* d_out, int out_size, void* d_ws, size_t ws_size,
                              hipStream_t stream) {
  const float* x      = (const float*)d_in[0];
  const float* w1     = (const float*)d_in[1];
  const float* b1     = (const float*)d_in[2];
  const float* rW     = (const float*)d_in[3];
  const float* sW     = (const float*)d_in[4];
  const float* Wv     = (const float*)d_in[5];
  const float* Wq     = (const float*)d_in[6];
  const float* Wk     = (const float*)d_in[7];
  const float* Wo     = (const float*)d_in[8];
  const float* Wskip  = (const float*)d_in[9];
  const float* gw     = (const float*)d_in[10];
  const float* gb     = (const float*)d_in[11];
  const float* w1o    = (const float*)d_in[12];
  const float* b1o    = (const float*)d_in[13];
  const float* roW    = (const float*)d_in[14];
  const float* WvO    = (const float*)d_in[15];
  const float* WqO    = (const float*)d_in[16];
  const float* WkO    = (const float*)d_in[17];
  const float* WoO    = (const float*)d_in[18];
  const float* WskipO = (const float*)d_in[19];
  const float* Whid   = (const float*)d_in[20];
  const float* bhid   = (const float*)d_in[21];
  const float* Wout   = (const float*)d_in[22];
  const float* bout   = (const float*)d_in[23];
  const int*   srcI   = (const int*)d_in[24];
  const int*   dstI   = (const int*)d_in[25];
  float* out = (float*)d_out;

  char* ws = (char*)d_ws;
  size_t off = 0;
  auto alloc = [&](size_t bytes) -> char* {
    char* p = ws + off;
    off = (off + bytes + 255) & ~(size_t)255;
    return p;
  };
  float*    ysph   = (float*)alloc((size_t)E_ * 16 * 4);
  _Float16* rbf16  = (_Float16*)alloc((size_t)E_ * 16 * 2);
  _Float16* h16A   = (_Float16*)alloc((size_t)N_ * 512 * 2);
  _Float16* h16B   = (_Float16*)alloc((size_t)N_ * 512 * 2);
  float*    q      = (float*)alloc((size_t)N_ * 32 * 4);
  float*    agg    = (float*)alloc((size_t)N_ * 512 * 4);
  float*    q64    = (float*)alloc((size_t)N_ * 64 * 4);
  float*    aggO   = (float*)alloc((size_t)N_ * 64 * 4);
  float*    hout   = (float*)alloc((size_t)N_ * 64 * 4);
  _Float16* w1p    = (_Float16*)alloc(4 * 1024 * 2);
  _Float16* Wq16   = (_Float16*)alloc(4 * 1024 * 2);
  _Float16* Wk16   = (_Float16*)alloc(4 * 1024 * 2);
  _Float16* rW16   = (_Float16*)alloc(16 * 1024 * 2);
  _Float16* sW16   = (_Float16*)alloc(16 * 1024 * 2);
  _Float16* Wv16   = (_Float16*)alloc(16 * 1024 * 2);
  _Float16* Wo16   = (_Float16*)alloc(16 * 1024 * 2);
  _Float16* Wsk16  = (_Float16*)alloc(16 * 1024 * 2);
  _Float16* w1op   = (_Float16*)alloc(1024 * 2);
  _Float16* WqO16  = (_Float16*)alloc(2048 * 2);
  _Float16* WkO16  = (_Float16*)alloc(2048 * 2);
  _Float16* roW16  = (_Float16*)alloc(8192 * 2);
  _Float16* WvO16  = (_Float16*)alloc(8192 * 2);
  _Float16* WoO16  = (_Float16*)alloc(4096 * 2);
  _Float16* WskO16 = (_Float16*)alloc(2048 * 2);
  _Float16* Whid16 = (_Float16*)alloc(4096 * 2);
  _Float16* WoutP  = (_Float16*)alloc(1024 * 2);

  k_geom<<<(E_ + 255) / 256, 256, 0, stream>>>(x, srcI, dstI, ysph, rbf16);

  auto mkfrag = [&](_Float16* d, const float* s, int Kt, int ncols, int nmat) {
    int n = Kt * ncols * nmat;
    k_mk_bfrag<<<(n + 255) / 256, 256, 0, stream>>>(d, s, Kt, ncols, nmat);
  };
  mkfrag(Wq16, Wq, 32, 32, 4);     mkfrag(Wk16, Wk, 32, 32, 4);
  mkfrag(rW16, rW, 32, 32, 16);    mkfrag(sW16, sW, 32, 32, 16);
  mkfrag(Wv16, Wv, 32, 32, 16);    mkfrag(Wo16, Wo, 32, 32, 16);
  mkfrag(Wsk16, Wskip, 32, 32, 16);
  mkfrag(WqO16, WqO, 32, 64, 1);   mkfrag(WkO16, WkO, 32, 64, 1);
  mkfrag(roW16, roW, 32, 64, 4);   mkfrag(WvO16, WvO, 32, 64, 4);
  mkfrag(WoO16, WoO, 64, 64, 1);   mkfrag(WskO16, WskipO, 32, 64, 1);
  mkfrag(Whid16, Whid, 64, 64, 1);
  k_mk_bfrag_pad16<<<(4 * 1024 + 255) / 256, 256, 0, stream>>>(w1p, w1, 4);
  k_mk_bfrag_pad16<<<(1024 + 255) / 256, 256, 0, stream>>>(w1op, w1o, 1);
  k_mk_bfrag_wout<<<4, 256, 0, stream>>>(WoutP, Wout);
  k_zero16<<<(N_ * 512 + 255) / 256, 256, 0, stream>>>(h16A, N_ * 512);

  _Float16* hc = h16A;
  _Float16* hn = h16B;
  for (int i = 0; i < 4; ++i) {
    k_node_gemm_h0<<<N_ / 16, 32, 0, stream>>>(hc, Wq16 + i * 1024, 32, q);
    k_layer_dst<<<N_, 32, 0, stream>>>(hc, q, ysph, rbf16,
                                       w1p + i * 1024, b1 + i * 32, Wk16 + i * 1024,
                                       rW16 + i * 4096, sW16 + i * 4096, Wv16 + i * 4096, agg);
    k_node_update<<<N_ / 16, 32, 0, stream>>>(hc, agg, Wsk16 + i * 4096, Wo16 + i * 4096,
                                              gw + i * 128, gb + i * 128, hn);
    _Float16* t = hc; hc = hn; hn = t;
  }

  k_node_gemm_h0<<<N_ / 16, 32, 0, stream>>>(hc, WqO16, 64, q64);
  k_out_dst<<<N_, 32, 0, stream>>>(hc, q64, ysph, rbf16, w1op, b1o, WkO16, roW16, WvO16, aggO);
  k_hout<<<N_ / 16, 32, 0, stream>>>(aggO, hc, WoO16, WskO16, hout);
  k_pool_mlp<<<1, 32, 0, stream>>>(hout, Whid16, bhid, WoutP, bout, out);
}
